// RNN_40699110097800
// MI455X (gfx1250) — matrix operand order repaired and checked by reading, not compiled, not run on hardware
//
#include <hip/hip_runtime.h>
#include <hip/hip_bf16.h>

typedef __bf16 bf16;
typedef __attribute__((ext_vector_type(16))) __bf16 v16bf;
typedef __attribute__((ext_vector_type(8)))  __bf16 v8bf;
typedef __attribute__((ext_vector_type(8)))  float  v8f;
typedef __attribute__((ext_vector_type(4)))  int    i4v;

#define NWG   8
#define BLK   256
#define B_    32
#define H_    512
#define COLS  64            // H_ / NWG
#define DT_F        0.001f
#define DT_OVER_TAU 0.1f    // DT / TAU

// ---- gfx1250 async global->LDS path (guarded; falls back to uint4 copy) ----
#if defined(__has_builtin)
# if __has_builtin(__builtin_amdgcn_global_load_async_to_lds_b128)
#  define USE_ASYNC_LDS 1
# endif
# if __has_builtin(__builtin_amdgcn_s_wait_asynccnt)
#  define HAVE_WAIT_ASYNC_BUILTIN 1
# endif
#endif

__device__ __forceinline__ void wait_asynccnt0() {
#ifdef USE_ASYNC_LDS
# ifdef HAVE_WAIT_ASYNC_BUILTIN
  __builtin_amdgcn_s_wait_asynccnt(0);
# else
  asm volatile("s_wait_asynccnt 0x0" ::: "memory");
# endif
#endif
}

// ---------------- grid-wide sense-reversing barrier (8 co-resident WGs) ----
__device__ __forceinline__ void grid_sync(unsigned* cnt, unsigned* gen) {
  __threadfence();            // release: make this WG's global writes visible
  __syncthreads();
  if (threadIdx.x == 0) {
    unsigned g    = __hip_atomic_load(gen, __ATOMIC_RELAXED, __HIP_MEMORY_SCOPE_AGENT);
    unsigned prev = __hip_atomic_fetch_add(cnt, 1u, __ATOMIC_ACQ_REL, __HIP_MEMORY_SCOPE_AGENT);
    if (prev == NWG - 1u) {
      __hip_atomic_store(cnt, 0u,    __ATOMIC_RELAXED, __HIP_MEMORY_SCOPE_AGENT);
      __hip_atomic_store(gen, g + 1u, __ATOMIC_RELEASE, __HIP_MEMORY_SCOPE_AGENT);
    } else {
      while (__hip_atomic_load(gen, __ATOMIC_ACQUIRE, __HIP_MEMORY_SCOPE_AGENT) == g)
        __builtin_amdgcn_s_sleep(1);
    }
  }
  __syncthreads();
  __threadfence();            // acquire: invalidate stale near caches
}

__global__ void rnn_ws_init(unsigned* cnt) {
  if (threadIdx.x == 0) cnt[0] = 0u;   // gen may stay poisoned: barrier is generation-relative
}

// ---------------- persistent RNN kernel ------------------------------------
__global__ __launch_bounds__(BLK)
void rnn_persistent(const float* __restrict__ x,      // (B,T,1)
                    const float* __restrict__ h0,     // (B,H)
                    const float* __restrict__ p_relg, // (H)
                    const float* __restrict__ taug,   // (H)
                    const float* __restrict__ W_ih,   // (H,1)
                    const float* __restrict__ W_hh,   // (H,H)
                    const float* __restrict__ W_hz,   // (1,H)
                    const float* __restrict__ W_zh,   // (H,1)
                    const float* __restrict__ Wmask,  // (H,H)
                    float* __restrict__ out_z,        // (B,T)
                    float* __restrict__ out_h,        // (B,T,H)
                    bf16*  __restrict__ a_buf,        // [2][B][H] bf16 (ws)
                    float* __restrict__ zpart,        // [2][NWG][B] f32 (ws)
                    unsigned* cnt, unsigned* gen, int T) {
  __shared__ bf16  w_sh[COLS][H_];   // 64 KB : masked weight slice (bf16)
  __shared__ bf16  a_sh[B_][H_];     // 32 KB : full activation u⊙tanh(h)
  __shared__ float c_sh[B_][COLS];   //  8 KB : drive, then z-contributions
  __shared__ float x_sh[B_];
  __shared__ float z_sh[B_];

  const int wg   = blockIdx.x;
  const int tid  = threadIdx.x;
  const int j    = tid & 63;           // local hidden column (fixed per thread)
  const int brow = tid >> 6;           // 0..3
  const int jg   = wg * COLS + j;      // global hidden index

  // WMMA tile coordinates: 8 waves -> 2 (batch) x 4 (cols) tiles of 16x16
  const int wave = tid >> 5;
  const int lane = tid & 31;
  const int mt   = wave & 1;
  const int nt   = wave >> 1;
  const int lm   = lane & 15;
  const int lh   = lane >> 4;

  // ---- per-thread resident parameters & state (registers) ----
  const float pr  = p_relg[jg];
  const float itd = 1.0f / taug[jg];
  const float wih = W_ih[jg];
  const float wzh = W_zh[jg];
  const float whz = W_hz[jg];

  float h_r[8], u_r[8], t_r[8];
  #pragma unroll
  for (int i = 0; i < 8; ++i) {
    const int b = 4 * i + brow;
    float hv = h0[b * H_ + jg];
    float tv = tanhf(hv);
    h_r[i] = hv; t_r[i] = tv; u_r[i] = pr;
    a_buf[0 * B_ * H_ + b * H_ + jg] = (bf16)(pr * tv);   // a_0 = u0 ⊙ tanh(h0)
    c_sh[b][j] = tv * whz;                                 // contribution to z0
  }

  // ---- load masked weight slice into LDS as bf16 ----
  for (int e = tid; e < COLS * H_; e += BLK) {
    const int n = e >> 9;          // local row (output unit)
    const int k = e & (H_ - 1);
    const size_t gi = (size_t)(wg * COLS + n) * H_ + k;
    w_sh[n][k] = (bf16)(W_hh[gi] * Wmask[gi]);
  }
  __syncthreads();
  if (tid < B_) {                  // z0 partial for this WG
    float s = 0.f;
    for (int jj = 0; jj < COLS; ++jj) s += c_sh[tid][jj];
    zpart[0 * NWG * B_ + wg * B_ + tid] = s;
  }
  grid_sync(cnt, gen);             // publish a_0 / zpart_0 to all WGs

  // ---------------- time loop ----------------
  for (int t = 0; t < T; ++t) {
    const int rb = t & 1, wb = rb ^ 1;

    // pull full activation tile (bf16, 32 KB) into LDS
#ifdef USE_ASYNC_LDS
    {
      char* srcb = const_cast<char*>((const char*)(a_buf + rb * B_ * H_));
      char* dstb = (char*)&a_sh[0][0];
      #pragma unroll
      for (int i = 0; i < 8; ++i) {
        const int c = (tid + i * BLK) * 16;
        __builtin_amdgcn_global_load_async_to_lds_b128(
            (__attribute__((address_space(1))) i4v*)(void*)(srcb + c),
            (__attribute__((address_space(3))) i4v*)(void*)(dstb + c),
            0, 0);
      }
    }
#else
    {
      const uint4* src = (const uint4*)(a_buf + rb * B_ * H_);
      uint4*       dst = (uint4*)&a_sh[0][0];
      #pragma unroll
      for (int i = 0; i < (B_ * H_ * 2 / 16) / BLK; ++i)
        dst[tid + i * BLK] = src[tid + i * BLK];
    }
#endif
    if (tid < B_) {                // z_{t-1} = sum of WG partials; stage x_t
      const int b = tid;
      float s = 0.f;
      #pragma unroll
      for (int w = 0; w < NWG; ++w) s += zpart[rb * NWG * B_ + w * B_ + b];
      z_sh[b] = s;
      x_sh[b] = x[b * T + t];
      if (wg == 0 && t > 0) out_z[b * T + (t - 1)] = s;
    }
    wait_asynccnt0();              // all of this wave's async LDS writes landed
    __syncthreads();

    // ---- drive = a @ W_slice^T via v_wmma_f32_16x16x32_bf16 ----
    v8f acc = {};
    const bf16* arow = &a_sh[mt * 16 + lm][0];
    const bf16* wrow = &w_sh[nt * 16 + lm][0];
    #pragma unroll
    for (int kb = 0; kb < H_; kb += 32) {
      union { v16bf v; struct { v8bf lo, hi; } p; } A;
      A.p.lo = *(const v8bf*)(arow + kb + lh * 8);        // K = kb+lh*8 .. +7
      A.p.hi = *(const v8bf*)(arow + kb + lh * 8 + 16);   // K = +16 ..
      union { v16bf v; struct { v8bf lo, hi; } p; } Bu;
      Bu.p.lo = *(const v8bf*)(wrow + kb + lh * 8);       // K = kb+lh*8 .. +7
      Bu.p.hi = *(const v8bf*)(wrow + kb + lh * 8 + 16);  // K = +16 ..
      v16bf Bf = Bu.v;
      acc = __builtin_amdgcn_wmma_f32_16x16x32_bf16(
          false, A.v, false, Bf, (short)0, acc, false, false);
    }
    #pragma unroll
    for (int v = 0; v < 8; ++v)
      c_sh[mt * 16 + lh * 8 + v][nt * 16 + lm] = acc[v];
    __syncthreads();

    // ---- elementwise dynamics (registers) ----
    #pragma unroll
    for (int i = 0; i < 8; ++i) {
      const int b = 4 * i + brow;
      const float drive = c_sh[b][j];
      const float du = (pr - u_r[i]) * itd - pr * u_r[i] * t_r[i];
      const float un = u_r[i] + du * DT_F;
      const float hn = h_r[i] +
          (drive + x_sh[b] * wih + z_sh[b] * wzh - h_r[i]) * DT_OVER_TAU;
      const float tn = tanhf(hn);
      out_h[(size_t)(b * T + t) * H_ + jg] = hn;
      a_buf[wb * B_ * H_ + b * H_ + jg] = (bf16)(un * tn);  // a_{t+1}
      c_sh[b][j] = tn * whz;                                 // z_t contribution
      u_r[i] = un; h_r[i] = hn; t_r[i] = tn;
    }
    __syncthreads();

    if (tid < B_) {                // z_t partial for this WG
      float s = 0.f;
      for (int jj = 0; jj < COLS; ++jj) s += c_sh[tid][jj];
      zpart[wb * NWG * B_ + wg * B_ + tid] = s;
    }
    grid_sync(cnt, gen);           // publish a_{t+1}, zpart(z_t)
  }

  // final output z_{T-1}
  if (wg == 0 && tid < B_) {
    const int rb = T & 1;
    float s = 0.f;
    #pragma unroll
    for (int w = 0; w < NWG; ++w) s += zpart[rb * NWG * B_ + w * B_ + tid];
    out_z[tid * T + (T - 1)] = s;
  }
}

// ---------------- host launcher --------------------------------------------
extern "C" void kernel_launch(void* const* d_in, const int* in_sizes, int n_in,
                              void* d_out, int out_size, void* d_ws, size_t ws_size,
                              hipStream_t stream) {
  (void)n_in; (void)out_size; (void)ws_size;
  const float* x     = (const float*)d_in[0];
  const float* h0    = (const float*)d_in[1];
  const float* p_rel = (const float*)d_in[2];
  const float* tdep  = (const float*)d_in[3];
  const float* W_ih  = (const float*)d_in[4];
  const float* W_hh  = (const float*)d_in[5];
  const float* W_hz  = (const float*)d_in[6];
  const float* W_zh  = (const float*)d_in[7];
  const float* Wmask = (const float*)d_in[8];

  const int T = in_sizes[0] / B_;              // x is (B, T, 1)

  float* out_z = (float*)d_out;                // (B, T) then (B, T, H)
  float* out_h = (float*)d_out + (size_t)B_ * T;

  // workspace layout: [0,8) barrier | 256.. a_buf (2*B*H bf16) | zpart (2*NWG*B f32)
  char* ws = (char*)d_ws;
  unsigned* cnt = (unsigned*)ws;
  unsigned* gen = (unsigned*)(ws + 4);
  bf16*  a_buf  = (bf16*)(ws + 256);
  float* zpart  = (float*)(ws + 256 + 2 * B_ * H_ * sizeof(bf16));

  rnn_ws_init<<<1, 32, 0, stream>>>(cnt);
  rnn_persistent<<<dim3(NWG), dim3(BLK), 0, stream>>>(
      x, h0, p_rel, tdep, W_ih, W_hh, W_hz, W_zh, Wmask,
      out_z, out_h, a_buf, zpart, cnt, gen, T);
}
